// ResidualDCN_62637803045633
// MI455X (gfx1250) — matrix operand order repaired and checked by reading, not compiled, not run on hardware
//
#include <hip/hip_runtime.h>

// ---------------------------------------------------------------------------
// ResidualDCN for MI455X (gfx1250, wave32, WMMA).
//
// Both convs: out[o,l] = sum_{c,k} W[o,c,k] * S[c,k,l], reduction depth 352.
// The reduction is iterated K-MAJOR (ck' = k*32 + c), so each 32-wide
// v_wmma_f32_16x16x32_f16 K-step is exactly one tap k=q:
//   - one ds_load_b128 sampling-table entry per chunk (shared by all 16 h),
//   - per-h gather offset h*L is a compile-time immediate in global_load_b32,
//   - weights are staged into LDS (f16) already permuted to k-major.
// Bilinear-sample validity is folded into the lerp weights (clamp + masked
// weights), so every gather is an unconditional in-bounds load: no execz
// branches.  act (32MB) lives in the 192MB L2, so gathers are L2 hits.
// A-fragments use contiguous k order (k = 16*hiHalf + s), same as B: two
// ds_load_b128 per fragment.
// ---------------------------------------------------------------------------

typedef __attribute__((ext_vector_type(16))) _Float16 v16h;
typedef __attribute__((ext_vector_type(8)))  _Float16 v8h;
typedef __attribute__((ext_vector_type(8)))  float    v8f;

#define B_        8
#define C_        32
#define L_        32768
#define K_        11
#define DIL_      10
#define PAD_SAME_ 50          // ceil(10*(11-1)/2)
#define PAD_OFF_  5           // (11-1)/2
#define CK_       (C_ * K_)   // 352 = reduction depth
#define NCHUNK_   K_          // k-major: 11 chunks of 32 channels
#define TILES_PER_BLOCK 8
#define THREADS_  (TILES_PER_BLOCK * 32)
#define NTILES_   (B_ * (L_ / 16))   // 16384 (b, l-tile) tiles

// ------------------------------ BN + ReLU ---------------------------------
__global__ __launch_bounds__(256)
void bn_relu_kernel(const float* __restrict__ in,
                    const float* __restrict__ gamma,
                    const float* __restrict__ beta,
                    const float* __restrict__ mean,
                    const float* __restrict__ var,
                    float* __restrict__ out) {
  const int i4 = (blockIdx.x * 256 + threadIdx.x) * 4;    // L_ % 4 == 0
  const int c  = (i4 >> 15) & (C_ - 1);                   // L_ = 2^15
  const float sc = gamma[c] * rsqrtf(var[c] + 1e-3f);
  const float mu = mean[c];
  const float bt = beta[c];
  const float4 v = *(const float4*)(in + i4);
  float4 r;
  r.x = fmaxf((v.x - mu) * sc + bt, 0.0f);
  r.y = fmaxf((v.y - mu) * sc + bt, 0.0f);
  r.z = fmaxf((v.z - mu) * sc + bt, 0.0f);
  r.w = fmaxf((v.w - mu) * sc + bt, 0.0f);
  *(float4*)(out + i4) = r;
}

// --------------------- generic WMMA (deformable) conv ---------------------
// A-fragment 16x32 f16 from LDS.  Contiguous layout: lane<16 holds row M=lane with
// K={0..15}; lane>=16 holds M=lane-16 with K={16..31}.
__device__ inline v16h load_a_frag(const _Float16* s_w, int row, int q, int hiHalf) {
  const _Float16* p = s_w + row * CK_ + q * 32 + hiHalf * 16;  // 16B aligned
  v8h lo = *(const v8h*)p;          // K = base .. base+7
  v8h hi = *(const v8h*)(p + 8);    // K = base+8 .. base+15
  v16h a;
#pragma unroll
  for (int i = 0; i < 8; ++i) { a[i] = lo[i]; a[8 + i] = hi[i]; }
  return a;
}

template <int NM, int COUT, int DIL, int PAD, bool TRANS, bool HAS_OFF, bool HAS_RES>
__global__ __launch_bounds__(THREADS_)
void wmma_conv_kernel(const float* __restrict__ act,      // [B, C, L]
                      const float* __restrict__ offsets,  // [B, L, K]
                      const float* __restrict__ w,        // [Cout, C, K]
                      const float* __restrict__ bias,     // [Cout]
                      const float* __restrict__ resid,    // [B, Cout, L]
                      float* __restrict__ out) {
  // f16 weights permuted to k-major (ck' = k*32 + c); rows COUT..NM*16 zero.
  __shared__ __align__(16) _Float16 s_w[NM * 16 * CK_];
  // per-wave sampling table: (column n, tap k) -> {loIdx, hiIdx, wLo, wHi}
  __shared__ __align__(16) int4 s_tab[TILES_PER_BLOCK][16 * K_];

  const int tid    = threadIdx.x;
  const int wave   = tid >> 5;
  const int lane   = tid & 31;
  const int laneN  = lane & 15;        // B/C/D column, A row-within-tile
  const int hiHalf = lane >> 4;        // 0: lanes 0-15, 1: lanes 16-31

  // --- stage permuted weights into LDS as f16 ---
  for (int j = tid; j < NM * 16 * CK_; j += THREADS_) {
    const int row = j / CK_;
    const int rem = j - row * CK_;
    const int k   = rem >> 5;          // tap
    const int c   = rem & 31;          // channel
    s_w[j] = (_Float16)((row < COUT) ? w[row * CK_ + c * K_ + k] : 0.0f);
  }

  // --- this wave's tile: batch b, 16 output positions starting at l0 ---
  const int tile = blockIdx.x * TILES_PER_BLOCK + wave;   // < NTILES_ always
  const int b    = tile >> 11;                            // L_/16 = 2048 tiles/batch
  const int l0   = (tile & 2047) * 16;

  // --- sampling table: clamp indices, fold validity into lerp weights ---
  for (int e = lane; e < 16 * K_; e += 32) {
    const int n = e / K_;
    const int k = e - n * K_;
    const int l = l0 + n;
    float off = 0.0f;
    if (HAS_OFF) off = offsets[(b * L_ + l) * K_ + k];
    const float pos = (float)(l + k * DIL - PAD) + off;
    const float fl  = floorf(pos);
    const int   lo  = (int)fl;
    const float fr  = pos - fl;
    int4 t;
    t.x = min(max(lo, 0), L_ - 1);                        // clamped lo index
    t.y = min(max(lo + 1, 0), L_ - 1);                    // clamped hi index
    t.z = __float_as_int((lo >= 0  && lo < L_)     ? 1.0f - fr : 0.0f);
    t.w = __float_as_int((lo >= -1 && lo < L_ - 1) ? fr        : 0.0f);
    s_tab[wave][e] = t;
  }
  __syncthreads();

  // --- accumulators seeded with bias (row o = mt*16 + r + hiHalf*8) ---
  v8f acc0, acc1;
#pragma unroll
  for (int r = 0; r < 8; ++r) {
    const int o0 = r + hiHalf * 8;
    acc0[r] = bias[min(o0, COUT - 1)] * ((o0 < COUT) ? 1.0f : 0.0f);
    if (NM > 1) {
      const int o1 = 16 + r + hiHalf * 8;
      acc1[r] = bias[min(o1, COUT - 1)] * ((o1 < COUT) ? 1.0f : 0.0f);
    }
  }

  // gather base: this half-wave covers channels hiHalf*16 .. hiHalf*16+15
  const float* actb = act + b * (C_ * L_) + hiHalf * (16 * L_);

  // --- GEMM: 11 k-major chunks of 32 channels, one tap per chunk ---
#pragma unroll
  for (int q = 0; q < NCHUNK_; ++q) {
    const int4 t = s_tab[wave][laneN * K_ + q];           // ds_load_b128
    const float wLo = __int_as_float(t.z);
    const float wHi = __int_as_float(t.w);
    const float* pLo = actb + t.x;
    const float* pHi = actb + t.y;
    // B-fragment: bf[h] = sample(channel hiHalf*16+h, tap q, column laneN);
    // h*L_ is a compile-time immediate offset in each global_load_b32.
    v16h bf;
#pragma unroll
    for (int h = 0; h < 16; ++h) {
      const float v = pLo[h * L_] * wLo + pHi[h * L_] * wHi;
      bf[h] = (_Float16)v;
    }

    v16h a0 = load_a_frag(s_w, laneN, q, hiHalf);
    acc0 = __builtin_amdgcn_wmma_f32_16x16x32_f16(false, a0, false, bf,
                                                  (short)0, acc0, false, false);
    if (NM > 1) {
      v16h a1 = load_a_frag(s_w, 16 + laneN, q, hiHalf);
      acc1 = __builtin_amdgcn_wmma_f32_16x16x32_f16(false, a1, false, bf,
                                                    (short)0, acc1, false, false);
    }
  }

  // --- store D (fused transpose for offset conv, residual for final conv) ---
#pragma unroll
  for (int r = 0; r < 8; ++r) {
    const int n  = laneN;
    const int o0 = r + hiHalf * 8;
    if (TRANS) {
      if (o0 < COUT)
        out[(b * L_ + l0 + n) * COUT + o0] = acc0[r];
    } else {
      const int oi0 = (b * COUT + o0) * L_ + l0 + n;
      out[oi0] = acc0[r] + (HAS_RES ? resid[oi0] : 0.0f);
    }
    if (NM > 1) {   // deform path: COUT = 32, all rows valid, never transposed
      const int o1  = 16 + r + hiHalf * 8;
      const int oi1 = (b * COUT + o1) * L_ + l0 + n;
      out[oi1] = acc1[r] + (HAS_RES ? resid[oi1] : 0.0f);
    }
  }
}

// ------------------------------- launcher ----------------------------------
extern "C" void kernel_launch(void* const* d_in, const int* in_sizes, int n_in,
                              void* d_out, int out_size, void* d_ws, size_t ws_size,
                              hipStream_t stream) {
  // setup_inputs() order:
  const float* x         = (const float*)d_in[0];
  const float* bn1_gamma = (const float*)d_in[1];
  const float* bn1_beta  = (const float*)d_in[2];
  const float* bn1_mean  = (const float*)d_in[3];
  const float* bn1_var   = (const float*)d_in[4];
  const float* bn2_gamma = (const float*)d_in[5];
  const float* bn2_beta  = (const float*)d_in[6];
  const float* bn2_mean  = (const float*)d_in[7];
  const float* bn2_var   = (const float*)d_in[8];
  const float* off1_w    = (const float*)d_in[9];
  const float* off1_b    = (const float*)d_in[10];
  const float* off2_w    = (const float*)d_in[11];
  const float* off2_b    = (const float*)d_in[12];
  const float* dc1_w     = (const float*)d_in[13];
  const float* dc1_b     = (const float*)d_in[14];
  const float* dc2_w     = (const float*)d_in[15];
  const float* dc2_b     = (const float*)d_in[16];
  float* outp = (float*)d_out;

  // workspace carve: act [B,C,L] | off [B,L,K] | d1 [B,C,L]  (~76 MB)
  float* act = (float*)d_ws;
  float* off = act + (size_t)B_ * C_ * L_;
  float* d1  = off + (size_t)B_ * L_ * K_;

  const int bn_blocks   = (B_ * C_ * L_) / 4 / 256;       // 8192
  const int conv_blocks = NTILES_ / TILES_PER_BLOCK;       // 2048

  // ---- stage 1 ----
  bn_relu_kernel<<<bn_blocks, 256, 0, stream>>>(x, bn1_gamma, bn1_beta,
                                                bn1_mean, bn1_var, act);
  // offset conv: Cout=11 (1 M-tile), dil=1, pad=5, transposed store [B,L,11]
  wmma_conv_kernel<1, K_, 1, PAD_OFF_, true, false, false>
      <<<conv_blocks, THREADS_, 0, stream>>>(act, nullptr, off1_w, off1_b,
                                             nullptr, off);
  // deformable conv 1: Cout=32 (2 M-tiles), dil=10, pad=50
  wmma_conv_kernel<2, C_, DIL_, PAD_SAME_, false, true, false>
      <<<conv_blocks, THREADS_, 0, stream>>>(act, off, dc1_w, dc1_b,
                                             nullptr, d1);

  // ---- stage 2 ----
  bn_relu_kernel<<<bn_blocks, 256, 0, stream>>>(d1, bn2_gamma, bn2_beta,
                                                bn2_mean, bn2_var, act);
  wmma_conv_kernel<1, K_, 1, PAD_OFF_, true, false, false>
      <<<conv_blocks, THREADS_, 0, stream>>>(act, nullptr, off2_w, off2_b,
                                             nullptr, off);
  // deformable conv 2 with fused residual (+x) straight into d_out
  wmma_conv_kernel<2, C_, DIL_, PAD_SAME_, false, true, true>
      <<<conv_blocks, THREADS_, 0, stream>>>(act, off, dc2_w, dc2_b,
                                             x, outp);
}
